// Nessler2009_84164179132542
// MI455X (gfx1250) — matrix operand order repaired and checked by reading, not compiled, not run on hardware
//
#include <hip/hip_runtime.h>
#include <hip/hip_bf16.h>
#include <math.h>

// ---------------------------------------------------------------------------
// Nessler-2009 STDP loop for MI455X (gfx1250, wave32, WMMA).
// GEMMs use V_WMMA_F32_16X16X32_F16 with operands loaded DIRECTLY from global
// as aligned b128 vector loads: A matrices kept row-major (M x K), B matrices
// kept transposed (N x K) so each lane's WMMA register slice is a contiguous
// 16/32-byte run. No LDS staging, no barriers inside the GEMM k-loop.
// f16 is exact for all trace counts (<=200) and one-hot data.
// ---------------------------------------------------------------------------

typedef __attribute__((ext_vector_type(16))) _Float16 v16h;
typedef __attribute__((ext_vector_type(8)))  _Float16 v8h;
typedef __attribute__((ext_vector_type(8)))  float    v8f;

#define BATCH   256
#define TSTEP   200
#define FEAT    1568
#define OUTS    100
#define OPAD    128           // O padded to 8 WMMA N-tiles
#define SIGMA_W 10
#define LRATE   1e-3f

// --------------------------- WMMA helpers ----------------------------------

__device__ __forceinline__ v8f wmma16(v16h a, v16h b, v8f c) {
  // D = A(16x32 f16) * B(32x16 f16) + C(16x16 f32)
  return __builtin_amdgcn_wmma_f32_16x16x32_f16(false, a, false, b,
                                                (short)0, c, false, false);
}

__device__ __forceinline__ v16h mk16(v8h lo, v8h hi) {
  return __builtin_shufflevector(lo, hi, 0, 1, 2, 3, 4, 5, 6, 7,
                                 8, 9, 10, 11, 12, 13, 14, 15);
}

// A operand: row = &A[m][k0] (row-major M x K). a[i] = K 16h+i
//   ->  one contiguous 32-byte load (matches load_b).
__device__ __forceinline__ v16h load_a(const _Float16* __restrict__ row, int h) {
  return *(const v16h*)(row + 16 * h);
}

// B operand from transposed storage: row = &BT[n][k0] (N x K). Here b[i] =
// B[k=16h+i][n] = BT[n][16h+i]  ->  one contiguous 32-byte load.
__device__ __forceinline__ v16h load_b(const _Float16* __restrict__ row, int h) {
  return *(const v16h*)(row + 16 * h);
}

// Deterministic counter-based RNG (step, batch) -> U[0,1)
__device__ __forceinline__ float u01(unsigned t, unsigned b) {
  unsigned s = t * 0x9E3779B9u ^ (b * 0x85EBCA6Bu + 0xC2B2AE35u);
  s ^= s >> 16; s *= 0x7FEB352Du; s ^= s >> 15; s *= 0x846CA68Bu; s ^= s >> 16;
  return (float)(s >> 8) * (1.0f / 16777216.0f);
}

// --------------------------- kernels ---------------------------------------

// Initialize all workspace state from the (const) inputs.
__global__ void init_k(const float* __restrict__ Lin, const float* __restrict__ pin,
                       float* Lm, float* prior, _Float16* LT16, _Float16* ldT,
                       float* pcf, _Float16* pchT, _Float16* winT) {
  int i0 = blockIdx.x * blockDim.x + threadIdx.x;
  int stride = gridDim.x * blockDim.x;
  for (int i = i0; i < FEAT * OUTS; i += stride) Lm[i] = Lin[i];
  for (int i = i0; i < OPAD; i += stride) prior[i] = (i < OUTS) ? pin[i] : 0.0f;
  for (int i = i0; i < OPAD * FEAT; i += stride) {      // L^T, padded rows
    int o = i / FEAT, f = i % FEAT;
    LT16[i] = (_Float16)((o < OUTS) ? Lin[f * OUTS + o] : 0.0f);
  }
  for (int i = i0; i < FEAT * BATCH; i += stride) ldT[i] = (_Float16)0.0f;
  for (int i = i0; i < BATCH * OPAD; i += stride) pcf[i] = 0.0f;
  for (int i = i0; i < OPAD * BATCH; i += stride) {
    pchT[i] = (_Float16)0.0f; winT[i] = (_Float16)0.0f;
  }
}

// Per-step trace computation. ltp = window sum (<=10), pot = window OR,
// ltd = running prefix sum up to t-SIGMA (state RMW). All exact in f16.
__global__ void trace_k(const int* __restrict__ x, _Float16* ldT, _Float16* ltpT,
                        _Float16* xT, _Float16* pot, int t) {
  int idx = blockIdx.x * blockDim.x + threadIdx.x;
  if (idx >= BATCH * FEAT) return;
  int b = idx / FEAT, f = idx % FEAT;
  const int* xp = x + (size_t)b * TSTEP * FEAT + f;
  int lo = t - (SIGMA_W - 1); if (lo < 0) lo = 0;
  int s = 0;
  for (int u = lo; u <= t; ++u) s += (xp[(size_t)u * FEAT] > 0);
  _Float16 ld = ldT[f * BATCH + b];
  if (t >= SIGMA_W) ld = ld + (_Float16)((xp[(size_t)(t - SIGMA_W) * FEAT] > 0) ? 1.0f : 0.0f);
  ldT[f * BATCH + b]  = ld;
  ltpT[f * BATCH + b] = (_Float16)(float)s;
  xT[f * BATCH + b]   = (_Float16)((xp[(size_t)t * FEAT] > 0) ? 1.0f : 0.0f);
  pot[b * FEAT + f]   = (_Float16)((s > 0) ? 1.0f : 0.0f);
}

// e = pot(256x1568) @ L(1568x100) + prior, with L stored transposed (128x1568).
// 8 waves/block = the 8 N-tiles of one M-tile; grid = 16 M-tiles.
__global__ void fwd_gemm(const _Float16* __restrict__ pot,
                         const _Float16* __restrict__ LT16,
                         const float* __restrict__ prior, float* __restrict__ e) {
  int wave = threadIdx.x >> 5, lane = threadIdx.x & 31;
  int h = lane >> 4, mn = lane & 15;
  int m0 = blockIdx.x * 16;
  int n0 = wave * 16;
  const _Float16* __restrict__ arow = pot + (size_t)(m0 + mn) * FEAT;
  const _Float16* __restrict__ brow = LT16 + (size_t)(n0 + mn) * FEAT;
  v8f acc = {};
#pragma unroll 2
  for (int k0 = 0; k0 < FEAT; k0 += 32) {
    __builtin_prefetch(arow + k0 + 128, 0, 1);
    v16h a = load_a(arow + k0, h);
    v16h b = load_b(brow + k0, h);
    acc = wmma16(a, b, acc);
  }
  int oo = n0 + mn;
  float pr = (oo < OUTS) ? prior[oo] : 0.0f;
#pragma unroll
  for (int r = 0; r < 8; ++r) {
    int m = r + 8 * h;
    e[(m0 + m) * OPAD + oo] = acc[r] + pr;
  }
}

// Winner sampling + post-trace + prior update/normalize. Single block of 256.
// Writes winners / post_cum in TRANSPOSED (O_PAD x B) layout for the update GEMM.
__global__ void sample_k(const float* __restrict__ e, _Float16* winT, float* pcf,
                         _Float16* pchT, float* pm, float* prior, int t, float inv_decay) {
  __shared__ float sp[OPAD];
  __shared__ float slse;
  int b = threadIdx.x;
  if (b < OPAD) pm[b] = 0.0f;
  __syncthreads();
  // categorical(logits = exp(e))  =>  p ~ exp(exp(e))
  float tot = 0.0f;
  for (int o = 0; o < OUTS; ++o) tot += expf(expf(e[b * OPAD + o]));
  float u = u01((unsigned)t, (unsigned)b) * tot;
  int pick = OUTS - 1; float cum = 0.0f;
  for (int o = 0; o < OUTS; ++o) {
    cum += expf(expf(e[b * OPAD + o]));
    if (cum >= u) { pick = o; break; }
  }
  for (int o = 0; o < OPAD; ++o)
    winT[o * BATCH + b] = (_Float16)((o == pick) ? 1.0f : 0.0f);
  float pv = pcf[b * OPAD + pick] + 1.0f;
  pcf[b * OPAD + pick] = pv;
  pchT[pick * BATCH + b] = (_Float16)pv;
  atomicAdd(&pm[pick], 1.0f);
  __syncthreads();
  if (b < OUTS) {
    float pmv = pm[b] * (1.0f / (float)BATCH);
    float pr = prior[b];
    pr += ((5.0f * expf(-pr) - 1.0f) * pmv - (1.0f - pmv)) * (LRATE / inv_decay);
    sp[b] = fminf(0.0f, fmaxf(-5.0f, pr));
  }
  __syncthreads();
  if (b == 0) {
    float m = -1e30f;
    for (int o = 0; o < OUTS; ++o) m = fmaxf(m, sp[o]);
    float s = 0.0f;
    for (int o = 0; o < OUTS; ++o) s += expf(sp[o] - m);
    slse = m + logf(s);
  }
  __syncthreads();
  if (b < OUTS) prior[b] = sp[b] - slse;
}

// Fused update: accA=ltp^T@win, accD=ltd^T@win, accC=x^T@post_cum; then
// dw = ((5e^{-L}-1)*A/B + D/B - C/B) * LR/inv_decay applied to f32 master L.
// A matrices are (F x B) row-major; B matrices transposed (O_PAD x B).
__global__ void upd_gemm(const _Float16* __restrict__ ltpT, const _Float16* __restrict__ xT,
                         const _Float16* __restrict__ ldT, const _Float16* __restrict__ winT,
                         const _Float16* __restrict__ pchT, float* __restrict__ Lm, float scale) {
  int wave = threadIdx.x >> 5, lane = threadIdx.x & 31;
  int h = lane >> 4, mn = lane & 15;
  int f0 = blockIdx.x * 16;
  int n0 = wave * 16;
  const _Float16* __restrict__ aL = ltpT + (size_t)(f0 + mn) * BATCH;
  const _Float16* __restrict__ aD = ldT  + (size_t)(f0 + mn) * BATCH;
  const _Float16* __restrict__ aX = xT   + (size_t)(f0 + mn) * BATCH;
  const _Float16* __restrict__ bW = winT + (size_t)(n0 + mn) * BATCH;
  const _Float16* __restrict__ bP = pchT + (size_t)(n0 + mn) * BATCH;
  v8f accA = {}, accD = {}, accC = {};
#pragma unroll
  for (int k0 = 0; k0 < BATCH; k0 += 32) {
    __builtin_prefetch(aL + k0 + 128, 0, 1);
    v16h vW = load_b(bW + k0, h);
    v16h vP = load_b(bP + k0, h);
    accA = wmma16(load_a(aL + k0, h), vW, accA);
    accD = wmma16(load_a(aD + k0, h), vW, accD);
    accC = wmma16(load_a(aX + k0, h), vP, accC);
  }
  int oo = n0 + mn;
  if (oo < OUTS) {
    const float invB = 1.0f / (float)BATCH;
#pragma unroll
    for (int r = 0; r < 8; ++r) {
      int f = f0 + r + 8 * h;
      float Lv = Lm[f * OUTS + oo];
      // ref: dw = (coef*A - D - C)*s with D = (-ltd)^T w /B  =>  +ltd term
      float dw = ((5.0f * expf(-Lv) - 1.0f) * accA[r] * invB
                  + accD[r] * invB - accC[r] * invB) * scale;
      Lm[f * OUTS + oo] = Lv + dw;
    }
  }
}

// Pairwise (POP=2) clamp + logsumexp normalize; refresh transposed f16 L.
__global__ void norm_k(float* __restrict__ Lm, _Float16* __restrict__ LT16) {
  int i = blockIdx.x * blockDim.x + threadIdx.x;
  if (i >= (FEAT / 2) * OUTS) return;
  int f = i / OUTS, o = i % OUTS;
  float a = Lm[f * OUTS + o];
  float c = Lm[(f + FEAT / 2) * OUTS + o];
  a = fminf(0.0f, fmaxf(-5.0f, a));
  c = fminf(0.0f, fmaxf(-5.0f, c));
  float m = fmaxf(a, c);
  float lse = m + logf(expf(a - m) + expf(c - m));
  a -= lse; c -= lse;
  Lm[f * OUTS + o] = a;
  Lm[(f + FEAT / 2) * OUTS + o] = c;
  LT16[o * FEAT + f] = (_Float16)a;
  LT16[o * FEAT + f + FEAT / 2] = (_Float16)c;
}

__global__ void out_k(const float* __restrict__ Lm, const float* __restrict__ prior,
                      float* __restrict__ out) {
  int i = blockIdx.x * blockDim.x + threadIdx.x;
  if (i < FEAT * OUTS)             out[i] = Lm[i];
  else if (i < FEAT * OUTS + OUTS) out[i] = prior[i - FEAT * OUTS];
}

// --------------------------- launch ----------------------------------------

extern "C" void kernel_launch(void* const* d_in, const int* in_sizes, int n_in,
                              void* d_out, int out_size, void* d_ws, size_t ws_size,
                              hipStream_t stream) {
  (void)in_sizes; (void)n_in; (void)out_size; (void)ws_size;
  const int*   x   = (const int*)d_in[0];
  const float* Lin = (const float*)d_in[1];
  const float* pin = (const float*)d_in[2];
  float* out = (float*)d_out;

  char* ws = (char*)d_ws;
  size_t off = 0;
  auto carve = [&](size_t bytes) -> char* {
    char* p = ws + off;
    off = (off + bytes + 255) & ~(size_t)255;
    return p;
  };
  float*    Lm    = (float*)   carve((size_t)FEAT * OUTS * 4);   // master L (f32)
  float*    prior = (float*)   carve((size_t)OPAD * 4);          // master prior
  _Float16* LT16  = (_Float16*)carve((size_t)OPAD * FEAT * 2);   // L^T f16 (OxF)
  _Float16* pot   = (_Float16*)carve((size_t)BATCH * FEAT * 2);  // pot (B x F)
  _Float16* ltpT  = (_Float16*)carve((size_t)FEAT * BATCH * 2);  // ltp^T (F x B)
  _Float16* xT    = (_Float16*)carve((size_t)FEAT * BATCH * 2);  // x^T
  _Float16* ldT   = (_Float16*)carve((size_t)FEAT * BATCH * 2);  // ltd^T (state)
  float*    e     = (float*)   carve((size_t)BATCH * OPAD * 4);  // membrane e
  _Float16* winT  = (_Float16*)carve((size_t)OPAD * BATCH * 2);  // winners^T
  float*    pcf   = (float*)   carve((size_t)BATCH * OPAD * 4);  // post_cum f32
  _Float16* pchT  = (_Float16*)carve((size_t)OPAD * BATCH * 2);  // post_cum^T f16
  float*    pm    = (float*)   carve((size_t)OPAD * 4);          // winner mean

  init_k<<<512, 256, 0, stream>>>(Lin, pin, Lm, prior, LT16, ldT, pcf, pchT, winT);

  for (int t = 0; t < TSTEP; ++t) {
    float inv_decay = 1.0f + (float)t;
    trace_k<<<(BATCH * FEAT) / 256, 256, 0, stream>>>(x, ldT, ltpT, xT, pot, t);
    fwd_gemm<<<BATCH / 16, 256, 0, stream>>>(pot, LT16, prior, e);
    sample_k<<<1, 256, 0, stream>>>(e, winT, pcf, pchT, pm, prior, t, inv_decay);
    upd_gemm<<<FEAT / 16, 256, 0, stream>>>(ltpT, xT, ldT, winT, pchT, Lm,
                                            LRATE / inv_decay);
    norm_k<<<((FEAT / 2) * OUTS + 255) / 256, 256, 0, stream>>>(Lm, LT16);
  }

  out_k<<<(FEAT * OUTS + OUTS + 255) / 256, 256, 0, stream>>>(Lm, prior, out);
}
